// RopeFlashAttention_78125455114814
// MI455X (gfx1250) — matrix operand order repaired and checked by reading, not compiled, not run on hardware
//
#include <hip/hip_runtime.h>
#include <hip/hip_bf16.h>
#include <cstdint>
#include <cstddef>

// ---------------------------------------------------------------------------
// MI455X (gfx1250) fused RoPE-attention block.
// bf16 WMMA (v_wmma_f32_16x16x32_bf16) for all GEMM-like math, fp32 accum.
// Double-buffered GLOBAL_LOAD_ASYNC_TO_LDS staging (ASYNCcnt) when available.
// ---------------------------------------------------------------------------

typedef __bf16 bf16;
typedef __attribute__((ext_vector_type(16))) __bf16 v16bf;
typedef __attribute__((ext_vector_type(8)))  __bf16 v8bf;
typedef __attribute__((ext_vector_type(4)))  __bf16 v4bf;
typedef __attribute__((ext_vector_type(8)))  float  v8f;

#if defined(__has_builtin)
#if __has_builtin(__builtin_amdgcn_global_load_async_to_lds_b128)
#define CDNA5_HAS_ASYNC 1
#endif
#endif

union FragBf { v16bf v; v8bf h[2]; };

__device__ __forceinline__ v8f wmma_bf16(v16bf a, v16bf b, v8f c) {
  // (neg_a, A, neg_b, B, c_mod, C, reuse_a, reuse_b)
  return __builtin_amdgcn_wmma_f32_16x16x32_bf16(false, a, false, b, (short)0, c,
                                                 false, false);
}

// 16B memory -> LDS copy. Async (no VGPR data, ASYNCcnt-tracked) on gfx1250.
__device__ __forceinline__ void cp_b128(void* lds, const void* g) {
#if defined(CDNA5_HAS_ASYNC)
  typedef int v4i __attribute__((vector_size(16)));
  typedef __attribute__((address_space(1))) v4i* gptr_t;  // global int4*
  typedef __attribute__((address_space(3))) v4i* lptr_t;  // LDS int4*
  __builtin_amdgcn_global_load_async_to_lds_b128((gptr_t)g, (lptr_t)lds, 0, 0);
#else
  *reinterpret_cast<v8bf*>(lds) = *reinterpret_cast<const v8bf*>(g);
#endif
}

__device__ __forceinline__ void wait_async0() {
#if defined(CDNA5_HAS_ASYNC)
#if __has_builtin(__builtin_amdgcn_s_wait_asynccnt)
  __builtin_amdgcn_s_wait_asynccnt(0);
#else
  asm volatile("s_wait_asynccnt 0" ::: "memory");
#endif
#endif
}

constexpr int CB  = 2;
constexpr int CS  = 2048;
constexpr int CD  = 4096;
constexpr int CH  = 32;
constexpr int CHD = 128;
constexpr int CTD = 3 * CD;      // 12288
constexpr int CM  = CB * CS;     // 4096 token rows

// ---------------------------------------------------------------------------
// fp32 -> bf16 elementwise (vectorized x4)
// ---------------------------------------------------------------------------
__global__ __launch_bounds__(256)
void k_cvt_bf16(const float* __restrict__ in, bf16* __restrict__ out, int n4) {
  const int i = blockIdx.x * 256 + threadIdx.x;
  if (i >= n4) return;
  const float4 f = reinterpret_cast<const float4*>(in)[i];
  v4bf o;
  o[0] = (bf16)f.x; o[1] = (bf16)f.y; o[2] = (bf16)f.z; o[3] = (bf16)f.w;
  reinterpret_cast<v4bf*>(out)[i] = o;
}

// ---------------------------------------------------------------------------
// fp32 [K][N] -> bf16 [N][K] tiled transpose (contraction-major weights)
// block (32,8), grid (N/32, K/32)
// ---------------------------------------------------------------------------
__global__ __launch_bounds__(256)
void k_transpose_bf16(const float* __restrict__ in, bf16* __restrict__ out,
                      int K, int N) {
  __shared__ float tile[32][33];
  const int kb = blockIdx.y * 32, nb = blockIdx.x * 32;
  const int tx = threadIdx.x, ty = threadIdx.y;
#pragma unroll
  for (int r = 0; r < 32; r += 8)
    tile[ty + r][tx] = in[(size_t)(kb + ty + r) * N + nb + tx];
  __syncthreads();
#pragma unroll
  for (int r = 0; r < 32; r += 8)
    out[(size_t)(nb + ty + r) * K + kb + tx] = (bf16)tile[tx][ty + r];
}

// ---------------------------------------------------------------------------
// bf16 GEMM: C[M][N] = A[M][K] * Bt[N][K]^T  (Bt is contraction-major)
// 256 threads = 8 waves (4x2), macro tile 128x128, K-step 32.
// Double-buffered async LDS staging. MODE 0: bf16 out; MODE 1: fp32 + bias.
// ---------------------------------------------------------------------------
template <int MODE>
__global__ __launch_bounds__(256)
void k_gemm_bf16(const bf16* __restrict__ A, const bf16* __restrict__ Bt,
                 bf16* __restrict__ outB, float* __restrict__ outF,
                 const float* __restrict__ bias, int M, int N, int K) {
  __shared__ __align__(16) bf16 As[2][128][48];  // 32 cols + pad, rows 96B
  __shared__ __align__(16) bf16 Bs[2][128][48];

  const int tid  = threadIdx.x;
  const int wave = tid >> 5, lane = tid & 31;
  const int lm = lane & 15, lh = lane >> 4;
  const int wm = (wave >> 1) * 32;   // wave M offset: 0..96
  const int wn = (wave & 1) * 64;    // wave N offset: 0/64
  const int bn = blockIdx.x * 128;
  const int bm = blockIdx.y * 128;

  const v8f vz = {0.f, 0.f, 0.f, 0.f, 0.f, 0.f, 0.f, 0.f};
  v8f acc[2][4];
#pragma unroll
  for (int i = 0; i < 2; ++i)
#pragma unroll
    for (int j = 0; j < 4; ++j) acc[i][j] = vz;

  const int r0 = tid >> 2;        // 0..63
  const int c0 = (tid & 3) * 8;   // 0,8,16,24

  auto stage = [&](int buf, int k0) {
#pragma unroll
    for (int c = 0; c < 2; ++c) {
      const int row = r0 + c * 64;
      cp_b128(&As[buf][row][c0], A + (size_t)(bm + row) * K + k0 + c0);
      cp_b128(&Bs[buf][row][c0], Bt + (size_t)(bn + row) * K + k0 + c0);
    }
  };

  stage(0, 0);
  for (int k0 = 0; k0 < K; k0 += 32) {
    const int cur = (k0 >> 5) & 1;
    wait_async0();        // tile `cur` landed in LDS (this wave's copies)
    __syncthreads();      // ...and everyone else's; prev compute done too
    if (k0 + 32 < K) {
      stage(cur ^ 1, k0 + 32);
      __builtin_prefetch(A + (size_t)(bm + r0) * K + k0 + 64 + c0, 0, 1);
    }

    // A fragments: lane=M row, K split {lh*8..+7, 16+lh*8..+7}
    FragBf af[2];
#pragma unroll
    for (int i = 0; i < 2; ++i) {
      const bf16* ar = &As[cur][wm + i * 16 + lm][0];
      af[i].h[0] = *reinterpret_cast<const v8bf*>(ar + lh * 8);
      af[i].h[1] = *reinterpret_cast<const v8bf*>(ar + 16 + lh * 8);
    }
    // B fragments: lane=N row of Bt, K split {lh*8..+7, 16+lh*8..+7}
#pragma unroll
    for (int j = 0; j < 4; ++j) {
      FragBf bfm;
      const bf16* br = &Bs[cur][wn + j * 16 + lm][0];
      bfm.h[0] = *reinterpret_cast<const v8bf*>(br + lh * 8);
      bfm.h[1] = *reinterpret_cast<const v8bf*>(br + 16 + lh * 8);
#pragma unroll
      for (int i = 0; i < 2; ++i)
        acc[i][j] = wmma_bf16(af[i].v, bfm.v, acc[i][j]);
    }
  }

  // Epilogue. C layout: vgpr r -> row r (lanes 0-15) / r+8 (lanes 16-31).
#pragma unroll
  for (int i = 0; i < 2; ++i)
#pragma unroll
    for (int j = 0; j < 4; ++j)
#pragma unroll
      for (int r = 0; r < 8; ++r) {
        const int row = bm + wm + i * 16 + r + 8 * lh;
        const int col = bn + wn + j * 16 + lm;
        const float v = acc[i][j][r];
        if (MODE == 0) outB[(size_t)row * N + col] = (bf16)v;
        else           outF[(size_t)row * N + col] = v + bias[col];
      }
}

// ---------------------------------------------------------------------------
// RoPE + repack: qkv[B*S][3*D] -> Q,K [B,H,S,Hd] (rotated), Vt [B,H,Hd,S]
// ---------------------------------------------------------------------------
__global__ __launch_bounds__(256)
void k_rope_pack(const bf16* __restrict__ qkv, bf16* __restrict__ Q,
                 bf16* __restrict__ Kc, bf16* __restrict__ Vt) {
  const int idx = blockIdx.x * 256 + threadIdx.x;
  const int j = idx & 63;
  const int h = (idx >> 6) & 31;
  const int s = (idx >> 11) & 2047;
  const int b = idx >> 22;

  const size_t rowoff = ((size_t)(b * CS + s)) * CTD + h * CHD + 2 * j;
  const float q0 = (float)qkv[rowoff];
  const float q1 = (float)qkv[rowoff + 1];
  const float k0 = (float)qkv[rowoff + CD];
  const float k1 = (float)qkv[rowoff + CD + 1];
  const float v0 = (float)qkv[rowoff + 2 * CD];
  const float v1 = (float)qkv[rowoff + 2 * CD + 1];

  // inv_freq = theta^(-2j/Hd); ln(10000) = 9.21034...
  const float inv_freq = __expf(-9.210340371976184f * (float)(2 * j) * (1.0f / 128.0f));
  const float ang = (float)s * inv_freq;
  float sn, cs;
  __sincosf(ang, &sn, &cs);

  const size_t qoff = (((size_t)(b * CH + h)) * CS + s) * CHD + 2 * j;
  Q[qoff]      = (bf16)(q0 * cs - q1 * sn);
  Q[qoff + 1]  = (bf16)(q0 * sn + q1 * cs);
  Kc[qoff]     = (bf16)(k0 * cs - k1 * sn);
  Kc[qoff + 1] = (bf16)(k0 * sn + k1 * cs);

  const size_t voff = (((size_t)(b * CH + h)) * CHD + 2 * j) * CS + s;
  Vt[voff]      = (bf16)v0;
  Vt[voff + CS] = (bf16)v1;
}

// ---------------------------------------------------------------------------
// Causal flash attention. Block = 4 waves; wave w owns 16 query rows.
// Double-buffered async K/V tiles; online softmax in WMMA C-register layout.
// grid (S/64, B*H), block 128.
// ---------------------------------------------------------------------------
__global__ __launch_bounds__(128)
void k_attn(const bf16* __restrict__ Q, const bf16* __restrict__ K,
            const bf16* __restrict__ Vt, bf16* __restrict__ ctx) {
  const int qtile = blockIdx.x;   // 0..31
  const int bh    = blockIdx.y;   // 0..63
  const int b = bh >> 5, h = bh & 31;
  const int tid  = threadIdx.x;
  const int wave = tid >> 5, lane = tid & 31;
  const int lm = lane & 15, lh = lane >> 4;

  __shared__ __align__(16) bf16 Ks[2][32][144];    // [s_row][hd] (+pad)
  __shared__ __align__(16) bf16 Vs[2][128][48];    // [hd][s_col] (+pad)
  __shared__ __align__(16) bf16 Ps[4][16][48];     // per-wave P tile [m][k]

  const bf16* Qh = Q  + (size_t)bh * CS * CHD;
  const bf16* Kh = K  + (size_t)bh * CS * CHD;
  const bf16* Vh = Vt + (size_t)bh * CHD * CS;

  const int q0 = qtile * 64 + wave * 16;

  // Q fragments (16 rows x 128 hd) live in VGPRs for the whole K loop.
  FragBf qf[4];
  {
    const bf16* qrow = Qh + (size_t)(q0 + lm) * CHD;
#pragma unroll
    for (int t = 0; t < 4; ++t) {
      qf[t].h[0] = *reinterpret_cast<const v8bf*>(qrow + t * 32 + lh * 8);
      qf[t].h[1] = *reinterpret_cast<const v8bf*>(qrow + t * 32 + 16 + lh * 8);
    }
  }

  const v8f vz = {0.f, 0.f, 0.f, 0.f, 0.f, 0.f, 0.f, 0.f};
  v8f O[8];
#pragma unroll
  for (int d = 0; d < 8; ++d) O[d] = vz;
  float mrow[8], lrow[8];
#pragma unroll
  for (int r = 0; r < 8; ++r) { mrow[r] = -3.0e38f; lrow[r] = 0.f; }

  auto stage_kv = [&](int buf, int kbase) {
#pragma unroll
    for (int c = 0; c < 4; ++c) {  // K tile: 32 x 128
      const int chunk = tid + c * 128;
      const int row = chunk >> 4, col = (chunk & 15) * 8;
      cp_b128(&Ks[buf][row][col], Kh + (size_t)(kbase + row) * CHD + col);
    }
#pragma unroll
    for (int c = 0; c < 4; ++c) {  // V tile: 128 x 32
      const int chunk = tid + c * 128;
      const int row = chunk >> 2, col = (chunk & 3) * 8;
      cp_b128(&Vs[buf][row][col], Vh + (size_t)row * CS + kbase + col);
    }
  };

  const float scale = 0.08838834764831845f;  // 1/sqrt(128)
  const int nkt = (qtile + 1) * 2;           // causal: keys up to qtile*64+63

  stage_kv(0, 0);
  for (int kt = 0; kt < nkt; ++kt) {
    const int kbase = kt * 32;
    const int cur = kt & 1;
    wait_async0();
    __syncthreads();
    if (kt + 1 < nkt) stage_kv(cur ^ 1, kbase + 32);

    if (kbase <= q0 + 15) {   // tile not fully masked for this wave
      // S = Q * K^T : two 16x16 score tiles, 4 WMMA k-steps over Hd each.
      v8f sa0 = vz, sa1 = vz;
#pragma unroll
      for (int t = 0; t < 4; ++t) {
        FragBf kb;
        kb.h[0] = *reinterpret_cast<const v8bf*>(&Ks[cur][lm][t * 32 + lh * 8]);
        kb.h[1] = *reinterpret_cast<const v8bf*>(&Ks[cur][lm][t * 32 + 16 + lh * 8]);
        sa0 = wmma_bf16(qf[t].v, kb.v, sa0);
      }
      const bool sub1 = (kbase + 16) <= (q0 + 15);
      if (sub1) {
#pragma unroll
        for (int t = 0; t < 4; ++t) {
          FragBf kb;
          kb.h[0] = *reinterpret_cast<const v8bf*>(&Ks[cur][16 + lm][t * 32 + lh * 8]);
          kb.h[1] = *reinterpret_cast<const v8bf*>(&Ks[cur][16 + lm][t * 32 + 16 + lh * 8]);
          sa1 = wmma_bf16(qf[t].v, kb.v, sa1);
        }
      }

      // Online softmax in C layout: row = q0 + r + 8*lh, col = kbase + {lm, 16+lm}.
#pragma unroll
      for (int r = 0; r < 8; ++r) {
        const int row = q0 + r + 8 * lh;
        const float s0 = sa0[r] * scale;
        const float s1 = sa1[r] * scale;
        const bool msk0 = (kbase + lm) > row;
        const bool msk1 = (!sub1) || ((kbase + 16 + lm) > row);
        float mx = fmaxf(msk0 ? -3.0e38f : s0, msk1 ? -3.0e38f : s1);
#pragma unroll
        for (int off = 1; off < 16; off <<= 1)
          mx = fmaxf(mx, __shfl_xor(mx, off, 32));
        const float mn = fmaxf(mrow[r], mx);
        const float p0 = msk0 ? 0.f : __expf(s0 - mn);
        const float p1 = msk1 ? 0.f : __expf(s1 - mn);
        float ps = p0 + p1;
#pragma unroll
        for (int off = 1; off < 16; off <<= 1)
          ps += __shfl_xor(ps, off, 32);
        const float alpha = __expf(mrow[r] - mn);
        lrow[r] = lrow[r] * alpha + ps;
        mrow[r] = mn;
#pragma unroll
        for (int d = 0; d < 8; ++d) O[d][r] = O[d][r] * alpha;
        // C-layout -> LDS (row-major P tile), reload below in A-layout.
        Ps[wave][r + 8 * lh][lm]      = (bf16)p0;
        Ps[wave][r + 8 * lh][16 + lm] = (bf16)p1;
      }

      FragBf pa;
      {
        const bf16* pr = &Ps[wave][lm][0];
        pa.h[0] = *reinterpret_cast<const v8bf*>(pr + lh * 8);
        pa.h[1] = *reinterpret_cast<const v8bf*>(pr + 16 + lh * 8);
      }
      // O += P * V : 8 head-dim tiles.
#pragma unroll
      for (int d = 0; d < 8; ++d) {
        FragBf vb;
        const bf16* vr = &Vs[cur][d * 16 + lm][0];
        vb.h[0] = *reinterpret_cast<const v8bf*>(vr + lh * 8);
        vb.h[1] = *reinterpret_cast<const v8bf*>(vr + 16 + lh * 8);
        O[d] = wmma_bf16(pa.v, vb.v, O[d]);
      }
    }
  }

  // Normalize and write ctx[b, s, h*Hd + hd] (bf16).
#pragma unroll
  for (int r = 0; r < 8; ++r) {
    const float il = 1.0f / lrow[r];
    const int srow = q0 + r + 8 * lh;
    bf16* crow = ctx + ((size_t)(b * CS + srow)) * CD + h * CHD;
#pragma unroll
    for (int d = 0; d < 8; ++d)
      crow[d * 16 + lm] = (bf16)(O[d][r] * il);
  }
}

// ---------------------------------------------------------------------------
extern "C" void kernel_launch(void* const* d_in, const int* in_sizes, int n_in,
                              void* d_out, int out_size, void* d_ws, size_t ws_size,
                              hipStream_t stream) {
  (void)in_sizes; (void)n_in; (void)out_size; (void)ws_size;
  const float* x      = (const float*)d_in[0];
  const float* w_qkv  = (const float*)d_in[1];
  const float* w_proj = (const float*)d_in[2];
  const float* b_proj = (const float*)d_in[3];
  float* out = (float*)d_out;

  char* ws = (char*)d_ws;
  size_t off = 0;
  auto take = [&](size_t bytes) -> char* {
    char* p = ws + off;
    off += (bytes + 255) & ~(size_t)255;
    return p;
  };
  bf16* xb    = (bf16*)take((size_t)CM * CD * 2);          // x bf16
  bf16* wqkvT = (bf16*)take((size_t)CTD * CD * 2);         // W_qkv^T bf16
  bf16* wprjT = (bf16*)take((size_t)CD * CD * 2);          // W_proj^T bf16
  bf16* qkv   = (bf16*)take((size_t)CM * CTD * 2);         // QKV bf16
  bf16* Qb    = (bf16*)take((size_t)CB * CH * CS * CHD * 2);
  bf16* Kb    = (bf16*)take((size_t)CB * CH * CS * CHD * 2);
  bf16* Vtb   = (bf16*)take((size_t)CB * CH * CHD * CS * 2);
  bf16* ctx   = (bf16*)take((size_t)CM * CD * 2);

  // 1) x -> bf16
  {
    const int n4 = CM * CD / 4;
    k_cvt_bf16<<<(n4 + 255) / 256, 256, 0, stream>>>(x, xb, n4);
  }
  // 2) weights -> bf16, transposed to contraction-major
  k_transpose_bf16<<<dim3(CTD / 32, CD / 32), dim3(32, 8), 0, stream>>>(w_qkv, wqkvT, CD, CTD);
  k_transpose_bf16<<<dim3(CD / 32, CD / 32), dim3(32, 8), 0, stream>>>(w_proj, wprjT, CD, CD);
  // 3) QKV projection (bf16 WMMA, fp32 accum, bf16 out)
  k_gemm_bf16<0><<<dim3(CTD / 128, CM / 128), 256, 0, stream>>>(
      xb, wqkvT, qkv, nullptr, nullptr, CM, CTD, CD);
  // 4) RoPE + head-major repack (V transposed for PV fragments)
  k_rope_pack<<<(CB * CS * CH * (CHD / 2)) / 256, 256, 0, stream>>>(qkv, Qb, Kb, Vtb);
  // 5) causal flash attention
  k_attn<<<dim3(CS / 64, CB * CH), 128, 0, stream>>>(Qb, Kb, Vtb, ctx);
  // 6) output projection + bias (fp32 out to d_out)
  k_gemm_bf16<1><<<dim3(CD / 128, CM / 128), 256, 0, stream>>>(
      ctx, wprjT, nullptr, out, b_proj, CM, CD, CD);
}
